// YOLOV3Loss_26268019982595
// MI455X (gfx1250) — compile-verified, matrix operand order checked by reading, not run on hardware
//
#include <hip/hip_runtime.h>
#include <hip/hip_bf16.h>

typedef __attribute__((ext_vector_type(16))) _Float16 v16h;
typedef __attribute__((ext_vector_type(8)))  float    v8f;

#define NB   32            // batch
#define NAN_ 12288         // anchors per image (A*G*G = 3*64*64)
#define NM   50            // max annotations
#define NC   80            // classes
#define TILE 256
#define NBLK (NAN_ / TILE) // 48

// Fused IoU-assignment + sparse BCE kernel. One thread per anchor.
// Deterministic block reduction done with a single v_wmma_f32_16x16x32_f16:
// A(16x32 f16) = [256 bce partials | 256 count partials], B = ones -> row sums.
__global__ __launch_bounds__(256) void yolo_iou_bce_kernel(
    const float* __restrict__ pred,    // (B, 12288, 85)
    const float* __restrict__ ann,     // (B, 50, 5)  cx,cy,w,h,cls
    const float* __restrict__ anc,     // (12288, 4)  cx,cy,w,h
    float* __restrict__ ws_bce,        // (B, NBLK)
    float* __restrict__ ws_cnt)        // (B, NBLK)
{
    __shared__ float sx1[NM], sy1[NM], sx2[NM], sy2[NM], sar[NM], scl[NM];
    __shared__ float red[512];

    const int tid = threadIdx.x;
    const int blk = blockIdx.x;
    const int b   = blockIdx.y;

    // Stage annotations for this image into LDS (corners + area + class).
    if (tid < NM) {
        const float* a = ann + ((size_t)b * NM + tid) * 5;
        float cx = a[0], cy = a[1], w = a[2], h = a[3], cl = a[4];
        sx1[tid] = cx - 0.5f * w;  sy1[tid] = cy - 0.5f * h;
        sx2[tid] = cx + 0.5f * w;  sy2[tid] = cy + 0.5f * h;
        sar[tid] = w * h;
        scl[tid] = cl;
    }
    __syncthreads();

    const int i = blk * TILE + tid;        // flat anchor index in [0, 12288)
    const float* av = anc + (size_t)i * 4;
    float ax1 = av[0] - 0.5f * av[2], ay1 = av[1] - 0.5f * av[3];
    float ax2 = av[0] + 0.5f * av[2], ay2 = av[1] + 0.5f * av[3];
    float area_a = (ax2 - ax1) * (ay2 - ay1);

    // Max-IoU assignment; strict '>' keeps first occurrence (jnp.argmax).
    float best = -2.0f; int arg = 0;
    #pragma unroll 5
    for (int m = 0; m < NM; ++m) {
        float iw = fmaxf(fminf(ax2, sx2[m]) - fmaxf(ax1, sx1[m]), 0.0f);
        float ih = fmaxf(fminf(ay2, sy2[m]) - fmaxf(ay1, sy1[m]), 0.0f);
        float inter = iw * ih;
        float iou = inter / fmaxf(area_a + sar[m] - inter, 1e-8f);
        iou = (scl[m] != -1.0f) ? iou : -1.0f;   // invalid ann never matches
        if (iou > best) { best = iou; arg = m; }
    }

    // Lazy BCE: only positive anchors ever touch the 133 MB prediction tensor.
    float bce = 0.0f, cnt = 0.0f;
    if (best >= 0.5f) {
        cnt = 1.0f;
        int k = (int)fminf(fmaxf(scl[arg], 0.0f), (float)(NC - 1));
        const float* p = pred + ((size_t)b * NAN_ + i) * (NC + 5) + 5;
        __builtin_prefetch(p, 0, 3);           // global_prefetch_b8
        float s = 0.0f;
        for (int c = 0; c < NC; ++c) {
            float pc = p[c];
            s -= (c == k) ? __logf(pc) : __logf(1.0f - pc);
        }
        bce = s;
    }

    // Stage partials: rows 0-7 of the A matrix = bce, rows 8-15 = counts.
    red[tid]       = bce;
    red[256 + tid] = cnt;
    __syncthreads();

    if (tid < 32) {                        // wave 0, EXEC all ones (wave32)
        // A fill, contiguous k order used here: lanes 0-15 -> M=lane, K=0..15;
        //                            lanes 16-31 -> M=lane-16, K=16..31.
        int m  = tid & 15;
        int k0 = (tid < 16) ? 0 : 16;
        v16h a;
        #pragma unroll
        for (int j = 0; j < 16; ++j) a[j] = (_Float16)red[m * 32 + k0 + j];
        v16h ones;
        #pragma unroll
        for (int j = 0; j < 16; ++j) ones[j] = (_Float16)1.0f;
        v8f c = {};
        // D[m][n] = sum_k A[m][k] : 16 row sums, broadcast across n.
        c = __builtin_amdgcn_wmma_f32_16x16x32_f16(
                /*neg_a=*/false, a, /*neg_b=*/false, ones,
                /*c_mod=*/(short)0, c, /*reuse_a=*/false, /*reuse_b=*/false);
        // C/D layout: lane 0 holds D[0..7][0] (bce rows), lane 16 holds
        // D[8..15][0] (count rows) in its 8 accumulator VGPRs.
        float s = c[0] + c[1] + c[2] + c[3] + c[4] + c[5] + c[6] + c[7];
        if (tid == 0)  ws_bce[b * NBLK + blk] = s;
        if (tid == 16) ws_cnt[b * NBLK + blk] = s;
    }
}

// One wave; one thread per image. Serial (deterministic) final reduction.
__global__ void yolo_finalize_kernel(
    const float* __restrict__ ws_bce,
    const float* __restrict__ ws_cnt,
    const float* __restrict__ ann,
    float* __restrict__ out)              // (2, B) row-major
{
    int b = threadIdx.x;
    if (b >= NB) return;
    float tb = 0.0f, tc = 0.0f;
    for (int j = 0; j < NBLK; ++j) {
        tb += ws_bce[b * NBLK + j];
        tc += ws_cnt[b * NBLK + j];
    }
    bool has = false;
    for (int m = 0; m < NM; ++m)
        has = has || (ann[((size_t)b * NM + m) * 5 + 4] != -1.0f);
    out[b]      = has ? (tb / fmaxf(tc, 1.0f)) : 0.0f;  // cls_loss row
    out[NB + b] = 0.0f;                                 // reg_loss row (no-op)
}

extern "C" void kernel_launch(void* const* d_in, const int* in_sizes, int n_in,
                              void* d_out, int out_size, void* d_ws, size_t ws_size,
                              hipStream_t stream) {
    (void)in_sizes; (void)n_in; (void)out_size; (void)ws_size;
    const float* pred = (const float*)d_in[0];  // (32,3,64,64,85)
    const float* ann  = (const float*)d_in[1];  // (32,50,5)
    const float* anc  = (const float*)d_in[2];  // (3,64,64,4)
    float* out = (float*)d_out;                 // (2,32)

    float* ws_bce = (float*)d_ws;               // 32*48 floats
    float* ws_cnt = ws_bce + NB * NBLK;         // 32*48 floats (12.3 KB total)

    dim3 grid(NBLK, NB);
    yolo_iou_bce_kernel<<<grid, TILE, 0, stream>>>(pred, ann, anc, ws_bce, ws_cnt);
    yolo_finalize_kernel<<<1, 32, 0, stream>>>(ws_bce, ws_cnt, ann, out);
}
